// ComplexLSTMModel_44770739093911
// MI455X (gfx1250) — matrix operand order repaired and checked by reading, not compiled, not run on hardware
//
#include <hip/hip_runtime.h>
#include <hip/hip_bf16.h>

typedef __attribute__((ext_vector_type(16))) __bf16        v16bf;
typedef __attribute__((ext_vector_type(8)))  float         v8f;
typedef __attribute__((ext_vector_type(4)))  unsigned int  v4u;

#define HID    512
#define GATES  2048
#define BATCH  256
#define SEQT   256
#define INDIM  128

// ---------- helpers ----------

__device__ __forceinline__ unsigned short f2bf(float f) {
    unsigned int u = __float_as_uint(f);
    u += 0x7FFFu + ((u >> 16) & 1u);           // round-to-nearest-even
    return (unsigned short)(u >> 16);
}

// A fragment (16x32 bf16, M x K): lane m = lane&15, half h = lane>>4.
// VGPR0-3 hold K = h*8 .. h*8+7 ; VGPR4-7 hold K = 16+h*8 .. 16+h*8+7.
// q must already point at (row m, k0 + h*8).
__device__ __forceinline__ v16bf loadA(const unsigned short* q) {
    union { v4u u[2]; v16bf v; } x;
    x.u[0] = *(const v4u*)(q);
    x.u[1] = *(const v4u*)(q + 16);
    return x.v;
}

// B fragment (32x16 bf16, K x N) fed from row-major W (N rows of length K):
// lane n = lane&15, half h = lane>>4. VGPR0-3 hold K = h*8 .. h*8+7 ; VGPR4-7 hold K = 16+h*8 .. 16+h*8+7.
// q must already point at (row n, k0 + h*8).
__device__ __forceinline__ v16bf loadB(const unsigned short* q) {
    union { v4u u[2]; v16bf v; } x;
    x.u[0] = *(const v4u*)(q);
    x.u[1] = *(const v4u*)(q + 16);
    return x.v;
}

__device__ __forceinline__ float sig_(float x)  { return 1.0f / (1.0f + __expf(-x)); }
__device__ __forceinline__ float tanh_(float x) { return 2.0f / (1.0f + __expf(-2.0f * x)) - 1.0f; }

// ---------- fused LSTM timestep ----------
// gates = x_t @ Wih^T + h_prev @ Whh^T + (bih+bhh);  cell update fused.
// Wave tile: 32 rows (2 M-tiles) x 16 h-cols x all 4 gates (8 accumulators),
// single software-pipelined K loop spanning the x-part then the h-part.
// grid = (HID/64, BATCH/32), block = 128 (4 waves).
template <int KX>
__global__ __launch_bounds__(128) void lstm_step_kernel(
    const unsigned short* __restrict__ xt,    // bf16 input at time t (row stride ldx)
    long                               ldx,
    const unsigned short* __restrict__ WihB,  // bf16 (GATES, KX) row-major
    const unsigned short* __restrict__ WhhB,  // bf16 (GATES, HID) row-major
    const float*          __restrict__ bsum,  // (GATES) = bih + bhh
    const unsigned short* __restrict__ hPrev, // bf16 (BATCH, HID)
    float*                __restrict__ cbuf,  // f32  (BATCH, HID)
    unsigned short*       __restrict__ hNext, // bf16 (BATCH, HID)
    unsigned short*       __restrict__ seqOut,// bf16 (BATCH, SEQT, HID) or null
    float*                __restrict__ hF,    // f32  (BATCH, HID) or null
    int t)
{
    const int lane = threadIdx.x & 31;
    const int wave = threadIdx.x >> 5;
    const int m0   = blockIdx.y * 32;             // 2 M-tiles of 16 rows
    const int c0   = blockIdx.x * 64 + wave * 16; // h-column tile (per gate)
    const int ml   = lane & 15;
    const int hh   = lane >> 4;

    constexpr int XI    = KX / 32;        // k-iters in the x part
    constexpr int ITERS = XI + HID / 32;  // total k-iters

    // fragment address generators (compile-time resolved under full unroll)
    auto aAddr = [&](int i, int mt) -> const unsigned short* {
        return (i < XI)
            ? xt    + (long)(m0 + mt * 16 + ml) * ldx + hh * 8 + i * 32
            : hPrev + (long)(m0 + mt * 16 + ml) * HID + hh * 8 + (i - XI) * 32;
    };
    auto bAddr = [&](int i, int g) -> const unsigned short* {
        return (i < XI)
            ? WihB + (long)(g * HID + c0 + ml) * KX  + hh * 8 + i * 32
            : WhhB + (long)(g * HID + c0 + ml) * HID + hh * 8 + (i - XI) * 32;
    };

    v8f acc[8] = {};   // [gate*2 + mtile]

    // software pipeline: current fragments in aC/bC, prefetch next into aN/bN
    v16bf aC[2], bC[4];
    aC[0] = loadA(aAddr(0, 0));
    aC[1] = loadA(aAddr(0, 1));
#pragma unroll
    for (int g = 0; g < 4; ++g) bC[g] = loadB(bAddr(0, g));

#pragma unroll
    for (int i = 0; i < ITERS; ++i) {
        const int in = (i + 1 < ITERS) ? i + 1 : i;   // last iter: dummy refetch
        v16bf aN[2], bN[4];
        aN[0] = loadA(aAddr(in, 0));
        aN[1] = loadA(aAddr(in, 1));
#pragma unroll
        for (int g = 0; g < 4; ++g) bN[g] = loadB(bAddr(in, g));

#pragma unroll
        for (int g = 0; g < 4; ++g) {
            acc[g * 2 + 0] = __builtin_amdgcn_wmma_f32_16x16x32_bf16(
                false, aC[0], false, bC[g], (short)0, acc[g * 2 + 0], false, false);
            acc[g * 2 + 1] = __builtin_amdgcn_wmma_f32_16x16x32_bf16(
                false, aC[1], false, bC[g], (short)0, acc[g * 2 + 1], false, false);
        }
        aC[0] = aN[0];
        aC[1] = aN[1];
#pragma unroll
        for (int g = 0; g < 4; ++g) bC[g] = bN[g];
    }

    // ---- fused LSTM cell epilogue ----
    // C/D layout: lane holds col N = ml; VGPR r holds row M = hh*8 + r.
    const int col = c0 + ml;
    const float bI = bsum[0 * HID + col];
    const float bF = bsum[1 * HID + col];
    const float bG = bsum[2 * HID + col];
    const float bO = bsum[3 * HID + col];
#pragma unroll
    for (int mt = 0; mt < 2; ++mt) {
#pragma unroll
        for (int r = 0; r < 8; ++r) {
            const int  row = m0 + mt * 16 + hh * 8 + r;
            const long ci  = (long)row * HID + col;
            float iv = sig_(acc[0 * 2 + mt][r] + bI);
            float fv = sig_(acc[1 * 2 + mt][r] + bF);
            float gv = tanh_(acc[2 * 2 + mt][r] + bG);
            float ov = sig_(acc[3 * 2 + mt][r] + bO);
            float cn = fv * cbuf[ci] + iv * gv;
            cbuf[ci] = cn;
            float hn = ov * tanh_(cn);
            hNext[ci] = f2bf(hn);
            if (seqOut) seqOut[((long)row * SEQT + t) * HID + col] = f2bf(hn);
            if (hF)     hF[ci] = hn;
        }
    }
}

// ---------- small utility kernels ----------

__global__ void f32_to_bf16_kernel(const float* __restrict__ in,
                                   unsigned short* __restrict__ out, long n) {
    long i = (long)blockIdx.x * blockDim.x + threadIdx.x;
    if (i < n) out[i] = f2bf(in[i]);
}

__global__ void bias_sum_kernel(const float* __restrict__ a, const float* __restrict__ b,
                                float* __restrict__ o, int n) {
    int i = blockIdx.x * blockDim.x + threadIdx.x;
    if (i < n) o[i] = a[i] + b[i];
}

__global__ void zero_b32_kernel(unsigned int* __restrict__ p, long n) {
    long i = (long)blockIdx.x * blockDim.x + threadIdx.x;
    if (i < n) p[i] = 0u;
}

// hid = relu(h_last @ W1^T + b1): (256,512) x (512,512)
__global__ void fc_relu_kernel(const float* __restrict__ x, const float* __restrict__ W,
                               const float* __restrict__ bias, float* __restrict__ y) {
    int idx = blockIdx.x * blockDim.x + threadIdx.x;
    int b = idx >> 9, j = idx & 511;
    const float* xr = x + (long)b * HID;
    const float* wr = W + (long)j * HID;
    float s = bias[j];
#pragma unroll 8
    for (int k = 0; k < HID; ++k) s += xr[k] * wr[k];
    y[idx] = fmaxf(s, 0.0f);
}

// out = hid @ W2^T + b2: (256,512) x (10,512)
__global__ void fc_out_kernel(const float* __restrict__ x, const float* __restrict__ W,
                              const float* __restrict__ bias, float* __restrict__ y, int n) {
    int idx = blockIdx.x * blockDim.x + threadIdx.x;
    if (idx >= n) return;
    int b = idx / 10, j = idx % 10;
    const float* xr = x + (long)b * HID;
    const float* wr = W + (long)j * HID;
    float s = bias[j];
#pragma unroll 8
    for (int k = 0; k < HID; ++k) s += xr[k] * wr[k];
    y[idx] = s;
}

// ---------- host launcher ----------

extern "C" void kernel_launch(void* const* d_in, const int* in_sizes, int n_in,
                              void* d_out, int out_size, void* d_ws, size_t ws_size,
                              hipStream_t stream) {
    const float* x    = (const float*)d_in[0];
    const float* Wih0 = (const float*)d_in[1];
    const float* Whh0 = (const float*)d_in[2];
    const float* bih0 = (const float*)d_in[3];
    const float* bhh0 = (const float*)d_in[4];
    const float* Wih1 = (const float*)d_in[5];
    const float* Whh1 = (const float*)d_in[6];
    const float* bih1 = (const float*)d_in[7];
    const float* bhh1 = (const float*)d_in[8];
    const float* W1   = (const float*)d_in[9];
    const float* b1   = (const float*)d_in[10];
    const float* W2   = (const float*)d_in[11];
    const float* b2   = (const float*)d_in[12];

    // workspace carve-up (256B aligned)
    char* w = (char*)d_ws;
    auto carve = [&](size_t bytes) -> void* {
        void* p = (void*)w;
        w += (bytes + 255) & ~(size_t)255;
        return p;
    };
    unsigned short* xB    = (unsigned short*)carve((size_t)BATCH * SEQT * INDIM * 2);
    unsigned short* seq0  = (unsigned short*)carve((size_t)BATCH * SEQT * HID * 2);
    unsigned short* Wih0B = (unsigned short*)carve((size_t)GATES * INDIM * 2);
    unsigned short* Whh0B = (unsigned short*)carve((size_t)GATES * HID * 2);
    unsigned short* Wih1B = (unsigned short*)carve((size_t)GATES * HID * 2);
    unsigned short* Whh1B = (unsigned short*)carve((size_t)GATES * HID * 2);
    float*          bsum0 = (float*)carve((size_t)GATES * 4);
    float*          bsum1 = (float*)carve((size_t)GATES * 4);
    unsigned short* hPing = (unsigned short*)carve((size_t)BATCH * HID * 2);
    unsigned short* hPong = (unsigned short*)carve((size_t)BATCH * HID * 2);
    float*          cbuf  = (float*)carve((size_t)BATCH * HID * 4);
    float*          hF    = (float*)carve((size_t)BATCH * HID * 4);
    float*          hid   = (float*)carve((size_t)BATCH * HID * 4);

    const int CT = 256;
    auto cvt = [&](const float* src, unsigned short* dst, long n) {
        f32_to_bf16_kernel<<<(int)((n + CT - 1) / CT), CT, 0, stream>>>(src, dst, n);
    };

    // one-time (per call) bf16 conversions
    cvt(x,    xB,    (long)BATCH * SEQT * INDIM);
    cvt(Wih0, Wih0B, (long)GATES * INDIM);
    cvt(Whh0, Whh0B, (long)GATES * HID);
    cvt(Wih1, Wih1B, (long)GATES * HID);
    cvt(Whh1, Whh1B, (long)GATES * HID);
    bias_sum_kernel<<<GATES / CT, CT, 0, stream>>>(bih0, bhh0, bsum0, GATES);
    bias_sum_kernel<<<GATES / CT, CT, 0, stream>>>(bih1, bhh1, bsum1, GATES);

    const dim3 sgrid(HID / 64, BATCH / 32);   // (8,8) blocks, 4 waves each
    const dim3 sblk(128);

    // ---- layer 0 ----
    zero_b32_kernel<<<(BATCH * HID / 2 + CT - 1) / CT, CT, 0, stream>>>((unsigned int*)hPing, BATCH * HID / 2);
    zero_b32_kernel<<<(BATCH * HID + CT - 1) / CT, CT, 0, stream>>>((unsigned int*)cbuf, BATCH * HID);
    for (int t = 0; t < SEQT; ++t) {
        unsigned short* hPrev = (t & 1) ? hPong : hPing;
        unsigned short* hNext = (t & 1) ? hPing : hPong;
        lstm_step_kernel<INDIM><<<sgrid, sblk, 0, stream>>>(
            xB + (long)t * INDIM, (long)SEQT * INDIM,
            Wih0B, Whh0B, bsum0, hPrev, cbuf, hNext,
            seq0, nullptr, t);
    }

    // ---- layer 1 ----
    zero_b32_kernel<<<(BATCH * HID / 2 + CT - 1) / CT, CT, 0, stream>>>((unsigned int*)hPing, BATCH * HID / 2);
    zero_b32_kernel<<<(BATCH * HID + CT - 1) / CT, CT, 0, stream>>>((unsigned int*)cbuf, BATCH * HID);
    for (int t = 0; t < SEQT; ++t) {
        unsigned short* hPrev = (t & 1) ? hPong : hPing;
        unsigned short* hNext = (t & 1) ? hPing : hPong;
        lstm_step_kernel<HID><<<sgrid, sblk, 0, stream>>>(
            seq0 + (long)t * HID, (long)SEQT * HID,
            Wih1B, Whh1B, bsum1, hPrev, cbuf, hNext,
            nullptr, hF, t);
    }

    // ---- classifier head ----
    fc_relu_kernel<<<(BATCH * HID) / CT, CT, 0, stream>>>(hF, W1, b1, hid);
    fc_out_kernel<<<(BATCH * 10 + CT - 1) / CT, CT, 0, stream>>>(hid, W2, b2, (float*)d_out, BATCH * 10);
}
